// AttentionBlock_36112085024770
// MI455X (gfx1250) — compile-verified, matrix operand order checked by reading, not run on hardware
//
#include <hip/hip_runtime.h>
#include <hip/hip_bf16.h>
#include <stdint.h>

typedef __attribute__((ext_vector_type(16))) __bf16        bf16x16;
typedef __attribute__((ext_vector_type(8)))  float         f32x8;
typedef __attribute__((ext_vector_type(8)))  unsigned short us8;

union FragU { bf16x16 v; us8 h[2]; };

__device__ __forceinline__ unsigned short f2bf(float f) {
  unsigned u = __float_as_uint(f);
  unsigned r = (u + 0x7FFFu + ((u >> 16) & 1u)) >> 16;  // round-to-nearest-even
  return (unsigned short)r;
}

// Load a 16-bit A/B WMMA fragment per CDNA5 ISA layout:
// lane -> M (A) or N (B), elements -> K in two contiguous 8-elem chunks
// at K = g*8 and K = 16 + g*8 (g = lane/16).
__device__ __forceinline__ bf16x16 ld_frag(const unsigned short* p, int stride, int row, int g) {
  const unsigned short* r = p + (size_t)row * stride + g * 8;
  FragU f;
  f.h[0] = *(const us8*)(r);
  f.h[1] = *(const us8*)(r + 16);
  return f.v;
}

__device__ __forceinline__ f32x8 wmma_bf16(bf16x16 a, bf16x16 b, f32x8 c) {
  return __builtin_amdgcn_wmma_f32_16x16x32_bf16(false, a, false, b, (short)0, c, false, false);
}

// CDNA5 async global->LDS DMA (ASYNCcnt-tracked), 16 bytes per lane.
__device__ __forceinline__ void async_b128(void* lds, const void* gptr) {
  unsigned ldsa = (unsigned)(unsigned long long)lds;   // addr[31:0] = LDS offset
  asm volatile("global_load_async_to_lds_b128 %0, %1, off"
               :: "v"(ldsa), "v"(gptr) : "memory");
}
__device__ __forceinline__ void wait_async0() {
  asm volatile("s_wait_asynccnt 0" ::: "memory");
}

// ---------------- fp32 -> bf16 convert ----------------
__global__ void cvt_kernel(const float* __restrict__ s, unsigned short* __restrict__ d, int n) {
  int i = blockIdx.x * 256 + threadIdx.x;
  if (i < n) d[i] = f2bf(s[i]);
}

// fp32 [batch][R][C] -> bf16 transposed [batch][C][R]
__global__ void cvt_t_kernel(const float* __restrict__ s, unsigned short* __restrict__ d,
                             int R, int C) {
  const int r = blockIdx.y;
  const int c = blockIdx.x * 256 + threadIdx.x;
  const size_t base = (size_t)blockIdx.z * R * C;
  if (c < C) d[base + (size_t)c * R + r] = f2bf(s[base + (size_t)r * C + c]);
}

// ---------------- LayerNorm (E = 1024) ----------------
__global__ __launch_bounds__(256)
void ln_kernel(const float* __restrict__ x,
               const float* __restrict__ gam,
               const float* __restrict__ bet,
               float* __restrict__ outF,
               unsigned short* __restrict__ outB) {
  __shared__ float2 red[256];
  const int row = blockIdx.x;
  const int t = threadIdx.x;
  const float* xr = x + (size_t)row * 1024;
  float vals[4];
  float s = 0.f, s2 = 0.f;
#pragma unroll
  for (int j = 0; j < 4; ++j) {
    float a = xr[t + 256 * j];
    vals[j] = a; s += a; s2 += a * a;
  }
  red[t] = make_float2(s, s2);
  __syncthreads();
  for (int off = 128; off > 0; off >>= 1) {
    if (t < off) { red[t].x += red[t + off].x; red[t].y += red[t + off].y; }
    __syncthreads();
  }
  const float mean = red[0].x * (1.f / 1024.f);
  const float var  = red[0].y * (1.f / 1024.f) - mean * mean;
  const float rstd = rsqrtf(var + 1e-5f);
#pragma unroll
  for (int j = 0; j < 4; ++j) {
    const int col = t + 256 * j;
    const float o = (vals[j] - mean) * rstd * gam[col] + bet[col];
    if (outF) outF[(size_t)row * 1024 + col] = o;
    if (outB) outB[(size_t)row * 1024 + col] = f2bf(o);
  }
}

// ---------------- Generic bf16 WMMA GEMM (B pre-transposed) ----------------
// C[M, 64-per-y-tile] = A[M,K] * B[K,N];  BT is B transposed: BT[n][k], ldbt = K.
// y-tile selects BT rows [y*64, y*64+64). Block tile 128x64, 8 waves of 32x32.
// Double-buffered LDS staging via async global->LDS DMA.
__global__ __launch_bounds__(256)
void gemm_bf16(const unsigned short* __restrict__ A, int lda,
               const unsigned short* __restrict__ BT, int ldbt,
               const float* __restrict__ bias,
               const float* __restrict__ resid,
               float* __restrict__ outF,
               unsigned short* __restrict__ outB,
               unsigned short* __restrict__ outBT,   // optional transposed bf16 out
               int ldc, int ldct, int K, int relu) {
  __shared__ unsigned short As[2][128][40];   // [m][k], 80B rows (16B aligned)
  __shared__ unsigned short Bs[2][64][40];    // [n][k]

  const int t    = threadIdx.x;
  const int lane = t & 31;
  const int wave = t >> 5;       // 0..7
  const int wm   = wave >> 1;    // 0..3 (M)
  const int wn   = wave & 1;     // 0..1 (N)
  const int g    = lane >> 4;
  const int ln16 = lane & 15;

  const size_t row0 = (size_t)blockIdx.x * 128;
  const int    n0   = blockIdx.y * 64;
  const unsigned short* Bg = BT + (size_t)blockIdx.y * 64 * ldbt;

  f32x8 acc[2][2];
#pragma unroll
  for (int i = 0; i < 2; ++i)
#pragma unroll
    for (int j = 0; j < 2; ++j) acc[i][j] = f32x8{0,0,0,0,0,0,0,0};

  const int arow = t >> 1;          // 0..127
  const int acol = (t & 1) * 16;    // 0 or 16
  const int brow = t >> 2;          // 0..63
  const int bcol = (t & 3) * 8;     // 0..24

  // stage tile (async DMA): A 128x32, BT 64x32
  auto stage = [&](int buf, int k0) {
    const unsigned short* ag = A + (row0 + arow) * lda + k0 + acol;
    async_b128(&As[buf][arow][acol],     ag);
    async_b128(&As[buf][arow][acol + 8], ag + 8);
    async_b128(&Bs[buf][brow][bcol], Bg + (size_t)brow * ldbt + k0 + bcol);
  };

  stage(0, 0);
  int buf = 0;
  for (int k0 = 0; k0 < K; k0 += 32, buf ^= 1) {
    wait_async0();        // this wave's DMA for buf complete
    __syncthreads();      // publish to all waves
    if (k0 + 32 < K) stage(buf ^ 1, k0 + 32);

    bf16x16 a0 = ld_frag(&As[buf][0][0], 40, wm * 32 + ln16, g);
    bf16x16 a1 = ld_frag(&As[buf][0][0], 40, wm * 32 + 16 + ln16, g);
    bf16x16 b0 = ld_frag(&Bs[buf][0][0], 40, wn * 32 + ln16, g);
    bf16x16 b1 = ld_frag(&Bs[buf][0][0], 40, wn * 32 + 16 + ln16, g);

    acc[0][0] = wmma_bf16(a0, b0, acc[0][0]);
    acc[0][1] = wmma_bf16(a0, b1, acc[0][1]);
    acc[1][0] = wmma_bf16(a1, b0, acc[1][0]);
    acc[1][1] = wmma_bf16(a1, b1, acc[1][1]);
    __syncthreads();      // all waves done with buf before it is re-staged
  }

  // epilogue: bias / relu / residual / store (f32, bf16, transposed bf16)
#pragma unroll
  for (int sm = 0; sm < 2; ++sm) {
#pragma unroll
    for (int sn = 0; sn < 2; ++sn) {
      const int col = n0 + wn * 32 + sn * 16 + ln16;
      const float bb = bias ? bias[col] : 0.0f;
#pragma unroll
      for (int i = 0; i < 8; ++i) {
        const size_t row = row0 + wm * 32 + sm * 16 + i + g * 8;
        float v = acc[sm][sn][i] + bb;
        if (relu) v = v > 0.0f ? v : 0.0f;
        if (resid) v += resid[row * ldc + col];
        if (outF)  outF[row * ldc + col] = v;
        if (outB)  outB[row * ldc + col] = f2bf(v);
        if (outBT) outBT[(size_t)col * ldct + row] = f2bf(v);
      }
    }
  }
}

// ---------------- Flash attention (causal), D = 64 ----------------
// q/k: bf16 [B*T, H*64] (head h at column h*64); vt: bf16 transposed [H*64][B*T].
// One wave per 16-row Q tile.
__global__ __launch_bounds__(32)
void attn_kernel(const unsigned short* __restrict__ q,
                 const unsigned short* __restrict__ k,
                 const unsigned short* __restrict__ vt,
                 const float* __restrict__ resid,
                 float* __restrict__ outF,
                 int T, int ldv, float scale) {
  __shared__ unsigned short pL[16][40];   // P tile row-major [m][s]

  const int lane = threadIdx.x;
  const int g    = lane >> 4;
  const int ln16 = lane & 15;
  const int m0   = blockIdx.x * 16;
  const int h    = blockIdx.y;
  const int b    = blockIdx.z;
  const size_t base  = (size_t)b * T * 1024 + h * 64;           // q/k row-major
  const unsigned short* vbase = vt + (size_t)h * 64 * ldv + (size_t)b * T;

  // Q fragments (16 x 64 split into two 16x32)
  bf16x16 qa0 = ld_frag(q + base + (size_t)m0 * 1024,      1024, ln16, g);
  bf16x16 qa1 = ld_frag(q + base + (size_t)m0 * 1024 + 32, 1024, ln16, g);

  f32x8 o[4];
#pragma unroll
  for (int j = 0; j < 4; ++j) o[j] = f32x8{0,0,0,0,0,0,0,0};
  float mrow[8], lrow[8];
#pragma unroll
  for (int i = 0; i < 8; ++i) { mrow[i] = -1e30f; lrow[i] = 0.f; }

  const int nchunks = m0 / 32 + 1;
  for (int c = 0; c < nchunks; ++c) {
    const int s0 = c * 32;

    float pe[2][8];
    // scores for two 16-column subtiles
#pragma unroll
    for (int st = 0; st < 2; ++st) {
      const int sb = s0 + st * 16;
      bf16x16 kb0 = ld_frag(k + base + (size_t)sb * 1024,      1024, ln16, g);
      bf16x16 kb1 = ld_frag(k + base + (size_t)sb * 1024 + 32, 1024, ln16, g);
      f32x8 s_acc = f32x8{0,0,0,0,0,0,0,0};
      s_acc = wmma_bf16(qa0, kb0, s_acc);
      s_acc = wmma_bf16(qa1, kb1, s_acc);
      const int col = sb + ln16;
#pragma unroll
      for (int i = 0; i < 8; ++i) {
        const int rowt = m0 + i + g * 8;
        float x = s_acc[i] * scale;
        pe[st][i] = (col > rowt) ? -1e30f : x;   // causal mask
      }
    }

    // online softmax update (row reductions across 16-lane halves)
#pragma unroll
    for (int i = 0; i < 8; ++i) {
      float m = fmaxf(pe[0][i], pe[1][i]);
      m = fmaxf(m, __shfl_xor(m, 1, 32));
      m = fmaxf(m, __shfl_xor(m, 2, 32));
      m = fmaxf(m, __shfl_xor(m, 4, 32));
      m = fmaxf(m, __shfl_xor(m, 8, 32));
      const float mn   = fmaxf(mrow[i], m);
      const float corr = __expf(mrow[i] - mn);
      mrow[i] = mn;
      lrow[i] *= corr;
#pragma unroll
      for (int j = 0; j < 4; ++j) o[j][i] *= corr;
      float e0 = __expf(pe[0][i] - mn);
      float e1 = __expf(pe[1][i] - mn);
      pe[0][i] = e0; pe[1][i] = e1;
      float s = e0 + e1;
      s += __shfl_xor(s, 1, 32);
      s += __shfl_xor(s, 2, 32);
      s += __shfl_xor(s, 4, 32);
      s += __shfl_xor(s, 8, 32);
      lrow[i] += s;
    }

    // P -> LDS (bf16, 16x32 row-major), re-gather as A fragment
#pragma unroll
    for (int st = 0; st < 2; ++st)
#pragma unroll
      for (int i = 0; i < 8; ++i)
        pL[i + g * 8][st * 16 + ln16] = f2bf(pe[st][i]);
    __syncthreads();

    bf16x16 pa = ld_frag(&pL[0][0], 40, ln16, g);
#pragma unroll
    for (int j = 0; j < 4; ++j) {
      // V^T fragment: lane -> d (N), elements -> s (K): contiguous global reads
      bf16x16 vb = ld_frag(vbase + s0, ldv, j * 16 + ln16, g);
      o[j] = wmma_bf16(pa, vb, o[j]);
    }
    __syncthreads();
  }

  // normalize + residual + store
#pragma unroll
  for (int i = 0; i < 8; ++i) {
    const float inv = 1.0f / lrow[i];
    const size_t row = (size_t)b * T + m0 + i + g * 8;
#pragma unroll
    for (int j = 0; j < 4; ++j) {
      const size_t idx = row * 1024 + h * 64 + j * 16 + ln16;
      outF[idx] = o[j][i] * inv + resid[idx];
    }
  }
}

// ---------------- Host orchestration ----------------
extern "C" void kernel_launch(void* const* d_in, const int* in_sizes, int n_in,
                              void* d_out, int out_size, void* d_ws, size_t ws_size,
                              hipStream_t stream) {
  (void)in_sizes; (void)n_in; (void)out_size; (void)ws_size;
  const int B = 2, T = 2048, E = 1024, H = 16, D = 64;
  const int N = B * T;        // 4096 rows
  const int FF = 4 * E;       // 4096
  const float scale = 1.0f / 32.0f;   // 1/sqrt(E)

  const float* x   = (const float*)d_in[0];
  const float* g1  = (const float*)d_in[1];  const float* b1  = (const float*)d_in[2];
  const float* wq1 = (const float*)d_in[3];  const float* wk1 = (const float*)d_in[4];
  const float* wv1 = (const float*)d_in[5];
  const float* g2  = (const float*)d_in[6];  const float* b2  = (const float*)d_in[7];
  const float* g3  = (const float*)d_in[8];  const float* b3  = (const float*)d_in[9];
  const float* wq2 = (const float*)d_in[10]; const float* wk2 = (const float*)d_in[11];
  const float* wv2 = (const float*)d_in[12];
  const float* g4  = (const float*)d_in[13]; const float* b4  = (const float*)d_in[14];
  const float* wl1 = (const float*)d_in[15]; const float* bl1 = (const float*)d_in[16];
  const float* wl2 = (const float*)d_in[17]; const float* bl2 = (const float*)d_in[18];
  const float* g5  = (const float*)d_in[19]; const float* b5  = (const float*)d_in[20];

  char* ws = (char*)d_ws;
  size_t off = 0;
  auto alloc = [&](size_t bytes) -> void* {
    void* p = ws + off;
    off += (bytes + 255) & ~(size_t)255;
    return p;
  };

  const size_t WSZ = (size_t)H * E * D;      // 1M elems per qkv weight
  // transposed bf16 weights: qkv [H][D][E], wl1t [FF][E], wl2t [E][FF]
  unsigned short* wq1t = (unsigned short*)alloc(WSZ * 2);
  unsigned short* wk1t = (unsigned short*)alloc(WSZ * 2);
  unsigned short* wv1t = (unsigned short*)alloc(WSZ * 2);
  unsigned short* wq2t = (unsigned short*)alloc(WSZ * 2);
  unsigned short* wk2t = (unsigned short*)alloc(WSZ * 2);
  unsigned short* wv2t = (unsigned short*)alloc(WSZ * 2);
  unsigned short* wl1t = (unsigned short*)alloc((size_t)E * FF * 2);
  unsigned short* wl2t = (unsigned short*)alloc((size_t)FF * E * 2);
  unsigned short* qb   = (unsigned short*)alloc((size_t)N * E * 2);
  unsigned short* kb   = (unsigned short*)alloc((size_t)N * E * 2);
  unsigned short* vtb  = (unsigned short*)alloc((size_t)N * E * 2);  // V transposed
  unsigned short* lnb  = (unsigned short*)alloc((size_t)N * E * 2);
  unsigned short* a1b  = (unsigned short*)alloc((size_t)N * FF * 2);
  float* rA = (float*)alloc((size_t)N * E * 4);
  float* rB = (float*)alloc((size_t)N * E * 4);
  float* rC = (float*)alloc((size_t)N * E * 4);

  auto cvt_t = [&](const float* s, unsigned short* d, int R, int C, int batch) {
    cvt_t_kernel<<<dim3((C + 255) / 256, R, batch), dim3(256), 0, stream>>>(s, d, R, C);
  };
  auto ln = [&](const float* in, const float* g, const float* bb,
                float* oF, unsigned short* oB) {
    ln_kernel<<<dim3(N), dim3(256), 0, stream>>>(in, g, bb, oF, oB);
  };
  auto gemm = [&](const unsigned short* A, int lda, const unsigned short* BT, int ldbt,
                  const float* bias, const float* resid, float* oF, unsigned short* oB,
                  unsigned short* oBT, int ldc, int ldct, int K, int ntiles, int relu) {
    gemm_bf16<<<dim3(N / 128, ntiles), dim3(256), 0, stream>>>(
        A, lda, BT, ldbt, bias, resid, oF, oB, oBT, ldc, ldct, K, relu);
  };
  auto attn = [&](const unsigned short* q_, const unsigned short* k_,
                  const unsigned short* vt_, const float* resid, float* oF) {
    attn_kernel<<<dim3(T / 16, H, B), dim3(32), 0, stream>>>(
        q_, k_, vt_, resid, oF, T, N, scale);
  };

  // weight conversion + transpose (bf16)
  cvt_t(wq1, wq1t, E, D, H); cvt_t(wk1, wk1t, E, D, H); cvt_t(wv1, wv1t, E, D, H);
  cvt_t(wq2, wq2t, E, D, H); cvt_t(wk2, wk2t, E, D, H); cvt_t(wv2, wv2t, E, D, H);
  cvt_t(wl1, wl1t, E, FF, 1); cvt_t(wl2, wl2t, FF, E, 1);

  // ---- block 1 ----
  ln(x, g1, b1, rA, lnb);                                              // lnb = LN1(x)
  gemm(lnb, E, wq1t, E, nullptr, nullptr, nullptr, qb, nullptr, E, 0, E, H, 0);
  gemm(lnb, E, wk1t, E, nullptr, nullptr, nullptr, kb, nullptr, E, 0, E, H, 0);
  gemm(lnb, E, wv1t, E, nullptr, nullptr, nullptr, nullptr, vtb, E, N, E, H, 0);
  attn(qb, kb, vtb, x, rB);                                            // rB = mha1 + x
  ln(rB, g2, b2, rA, nullptr);                                         // rA = y2
  // ---- block 2 ----
  ln(rA, g3, b3, nullptr, lnb);                                        // lnb = LN3(y2)
  gemm(lnb, E, wq2t, E, nullptr, nullptr, nullptr, qb, nullptr, E, 0, E, H, 0);
  gemm(lnb, E, wk2t, E, nullptr, nullptr, nullptr, kb, nullptr, E, 0, E, H, 0);
  gemm(lnb, E, wv2t, E, nullptr, nullptr, nullptr, nullptr, vtb, E, N, E, H, 0);
  attn(qb, kb, vtb, rA, rB);                                           // rB = mha2 + y2
  // ---- MLP ----
  ln(rB, g4, b4, rC, lnb);                                             // rC = z4 f32, lnb = z4 bf16
  gemm(lnb, E, wl1t, E, bl1, nullptr, nullptr, a1b, nullptr, FF, 0, E, FF / 64, 1);
  gemm(a1b, FF, wl2t, FF, bl2, rC, rB, nullptr, nullptr, E, 0, FF, E / 64, 0);
  ln(rB, g5, b5, (float*)d_out, nullptr);                              // final LN
}
